// T2GV2_noEnt_53919019434663
// MI455X (gfx1250) — compile-verified, matrix operand order checked by reading, not run on hardware
//
#include <hip/hip_runtime.h>
#include <hip/hip_bf16.h>
#include <math.h>

typedef __bf16 bf16_t;
typedef __bf16 v16bf __attribute__((ext_vector_type(16)));
typedef __bf16 v8bf  __attribute__((ext_vector_type(8)));
typedef float  v8f   __attribute__((ext_vector_type(8)));

// Problem constants
static constexpr int kD     = 768;
static constexpr int kN     = 10240;   // nodes
static constexpr int kE     = 163840;  // edges
static constexpr int kR     = 4;
static constexpr int kNCAT  = 3840;    // 4*768 rel + 768 root
static constexpr int kSEQ   = 320;     // B*EVI
static constexpr int kWORDS = 32;
static constexpr int kB     = 64;
static constexpr int kDCAT  = 2304;    // D + 2*DOUT

// ---------------------------------------------------------------------------
// WMMA bf16 GEMM: C[M,N] = A[M,K] @ B[K,N], fp32 accumulate.
// Block tile 128x128, BK=32, double-buffered LDS (one barrier per K-step).
// 8 waves; each wave owns a 32x64 patch = 2x4 WMMA fragments.
// ---------------------------------------------------------------------------
__device__ __forceinline__ v16bf mk16(v8bf lo, v8bf hi) {
  v16bf r;
#pragma unroll
  for (int i = 0; i < 8; ++i) { r[i] = lo[i]; r[i + 8] = hi[i]; }
  return r;
}

__global__ __launch_bounds__(256) void k_gemm_bf16(
    const bf16_t* __restrict__ A, const bf16_t* __restrict__ B,
    float* __restrict__ Cf, bf16_t* __restrict__ Cb,
    int M, int N, int K, int relu)
{
  constexpr int BM = 128, BN = 128, BK = 32, LDP = 40; // rows padded to 80B
  __shared__ bf16_t Alds[2][BM][LDP];
  __shared__ bf16_t Bt[2][BN][LDP];   // B tile transposed: [n][k]
  const int tid  = threadIdx.x;
  const int lane = tid & 31;
  const int wave = tid >> 5;
  const int wm   = wave & 3;       // 4 waves along M (32 rows each)
  const int wn   = wave >> 2;      // 2 waves along N (64 cols each)
  const int bm0  = blockIdx.x * BM;
  const int bn0  = blockIdx.y * BN;
  const int lm   = lane & 15;
  const int kb   = (lane >> 4) * 8; // 16-bit fragment layout K-base per half-wave
  const bool fullM = (bm0 + BM <= M);

  // Per-thread staging coordinates (2 chunks of 8 bf16 for A and for B).
  const int ar0 = tid >> 2,          ac0 = (tid & 3) << 3;
  const int ar1 = (tid + 256) >> 2,  ac1 = ((tid + 256) & 3) << 3;
  const int br0 = tid >> 4,          bc0 = (tid & 15) << 3;
  const int br1 = (tid + 256) >> 4,  bc1 = ((tid + 256) & 15) << 3;

  uint4 aR0, aR1, bR0, bR1;

  auto loadG = [&](int k0) {
    if (fullM) {
      aR0 = *(const uint4*)(A + (size_t)(bm0 + ar0) * K + k0 + ac0);
      aR1 = *(const uint4*)(A + (size_t)(bm0 + ar1) * K + k0 + ac1);
    } else {
      aR0 = uint4{0u, 0u, 0u, 0u};
      aR1 = uint4{0u, 0u, 0u, 0u};
      if (bm0 + ar0 < M) aR0 = *(const uint4*)(A + (size_t)(bm0 + ar0) * K + k0 + ac0);
      if (bm0 + ar1 < M) aR1 = *(const uint4*)(A + (size_t)(bm0 + ar1) * K + k0 + ac1);
    }
    bR0 = *(const uint4*)(B + (size_t)(k0 + br0) * N + bn0 + bc0);
    bR1 = *(const uint4*)(B + (size_t)(k0 + br1) * N + bn0 + bc1);
  };
  auto storeL = [&](int buf) {
    *(uint4*)(&Alds[buf][ar0][ac0]) = aR0;
    *(uint4*)(&Alds[buf][ar1][ac1]) = aR1;
    const bf16_t* p0 = (const bf16_t*)&bR0;
    const bf16_t* p1 = (const bf16_t*)&bR1;
#pragma unroll
    for (int j = 0; j < 8; ++j) Bt[buf][bc0 + j][br0] = p0[j];
#pragma unroll
    for (int j = 0; j < 8; ++j) Bt[buf][bc1 + j][br1] = p1[j];
  };

  v8f acc[2][4];
#pragma unroll
  for (int i = 0; i < 2; ++i)
#pragma unroll
    for (int j = 0; j < 4; ++j)
#pragma unroll
      for (int v = 0; v < 8; ++v) acc[i][j][v] = 0.f;

  const int nIter = K / BK;
  loadG(0);
  storeL(0);
  for (int it = 0; it < nIter; ++it) {
    const int buf = it & 1;
    if (it + 1 < nIter) loadG((it + 1) * BK);   // issue next tile's loads early
    __syncthreads();                            // buf is ready

    v16bf afr[2], bfr[4];
#pragma unroll
    for (int i = 0; i < 2; ++i) {
      int row = wm * 32 + i * 16 + lm;
      v8bf lo = *(const v8bf*)(&Alds[buf][row][kb]);
      v8bf hi = *(const v8bf*)(&Alds[buf][row][kb + 16]);
      afr[i] = mk16(lo, hi);
    }
#pragma unroll
    for (int j = 0; j < 4; ++j) {
      int cr = wn * 64 + j * 16 + lm;
      v8bf lo = *(const v8bf*)(&Bt[buf][cr][kb]);
      v8bf hi = *(const v8bf*)(&Bt[buf][cr][kb + 16]);
      bfr[j] = mk16(lo, hi);
    }
#pragma unroll
    for (int i = 0; i < 2; ++i)
#pragma unroll
      for (int j = 0; j < 4; ++j)
        acc[i][j] = __builtin_amdgcn_wmma_f32_16x16x32_bf16(
            false, afr[i], false, bfr[j], (short)0, acc[i][j], false, false);

    if (it + 1 < nIter) storeL((it + 1) & 1);   // drain loads into other buffer
  }

  // C/D layout: VGPR v, lane<16 -> (M=v, N=lane); lane>=16 -> (M=v+8, N=lane-16)
  const int ln   = lane & 15;
  const int rofs = (lane >> 4) * 8;
#pragma unroll
  for (int i = 0; i < 2; ++i) {
#pragma unroll
    for (int j = 0; j < 4; ++j) {
      int col = bn0 + wn * 64 + j * 16 + ln;
      int rb  = bm0 + wm * 32 + i * 16 + rofs;
#pragma unroll
      for (int v = 0; v < 8; ++v) {
        int row = rb + v;
        if (row < M) {
          float val = acc[i][j][v];
          if (relu) val = fmaxf(val, 0.f);
          if (Cf) Cf[(size_t)row * N + col] = val;
          if (Cb) Cb[(size_t)row * N + col] = (bf16_t)val;
        }
      }
    }
  }
}

// ---------------------------------------------------------------------------
// Small helper kernels
// ---------------------------------------------------------------------------
__global__ void k_gather_mean(const float* __restrict__ tok,
                              const int* __restrict__ wti,
                              bf16_t* __restrict__ xb) {
  int n = blockIdx.x, tid = threadIdx.x;
  int i0 = wti[n * 4 + 0], i1 = wti[n * 4 + 1];
  int i2 = wti[n * 4 + 2], i3 = wti[n * 4 + 3];
  for (int d = tid; d < kD; d += 256) {
    float s = tok[(size_t)i0 * kD + d] + tok[(size_t)i1 * kD + d] +
              tok[(size_t)i2 * kD + d] + tok[(size_t)i3 * kD + d];
    xb[(size_t)n * kD + d] = (bf16_t)(s * 0.25f);
  }
}

// pack [W_rel (4,768,768) | W_root (768,768)] -> bf16 (768, 3840)
__global__ void k_pack_wcat(const float* __restrict__ Wrel,
                            const float* __restrict__ Wroot,
                            bf16_t* __restrict__ out, int total) {
  int i = blockIdx.x * blockDim.x + threadIdx.x;
  if (i >= total) return;
  int d = i / kNCAT, c = i - d * kNCAT;
  int r = c / kD, e = c - r * kD;
  float v = (r < kR) ? Wrel[((size_t)r * kD + d) * kD + e]
                     : Wroot[(size_t)d * kD + e];
  out[i] = (bf16_t)v;
}

__global__ void k_f32_to_bf16(const float* __restrict__ in,
                              bf16_t* __restrict__ out, int total) {
  int i = blockIdx.x * blockDim.x + threadIdx.x;
  if (i < total) out[i] = (bf16_t)in[i];
}

__global__ void k_count(const int* __restrict__ ei, const int* __restrict__ et,
                        int* __restrict__ deg, int* __restrict__ cntDR) {
  int e = blockIdx.x * blockDim.x + threadIdx.x;
  if (e >= kE) return;
  int dst = ei[kE + e], t = et[e];
  atomicAdd(&deg[dst], 1);
  atomicAdd(&cntDR[dst * kR + t], 1);
}

__global__ __launch_bounds__(1024) void k_scan(const int* __restrict__ deg,
                                               int* __restrict__ off,
                                               int* __restrict__ cursor) {
  __shared__ int sh[1024];
  __shared__ int carry;
  int tid = threadIdx.x;
  if (tid == 0) carry = 0;
  __syncthreads();
  for (int base = 0; base < kN; base += 1024) {
    int i = base + tid;
    int v = (i < kN) ? deg[i] : 0;
    sh[tid] = v;
    __syncthreads();
    for (int s = 1; s < 1024; s <<= 1) {
      int t = (tid >= s) ? sh[tid - s] : 0;
      __syncthreads();
      sh[tid] += t;
      __syncthreads();
    }
    int inc = sh[tid];
    int c = carry;
    __syncthreads();
    if (tid == 1023) carry = c + sh[1023];
    int exc = c + inc - v;
    if (i < kN) { off[i] = exc; cursor[i] = exc; }
    __syncthreads();
  }
  if (tid == 0) off[kN] = carry;
}

__global__ void k_fill(const int* __restrict__ ei, const int* __restrict__ et,
                       int* __restrict__ cursor, int* __restrict__ elist) {
  int e = blockIdx.x * blockDim.x + threadIdx.x;
  if (e >= kE) return;
  int src = ei[e], dst = ei[kE + e], t = et[e];
  int pos = atomicAdd(&cursor[dst], 1);
  elist[pos] = (src << 2) | t;
}

__global__ void k_inv(const int* __restrict__ cntDR, float* __restrict__ invDR) {
  int i = blockIdx.x * blockDim.x + threadIdx.x;
  if (i < kN * kR) invDR[i] = 1.0f / fmaxf((float)cntDR[i], 1.0f);
}

// agg[dst] = sum_e inv[dst,t]*xr[t][src] + root + bias, relu. xrcat bf16 (N,3840).
__global__ __launch_bounds__(256) void k_aggregate(
    const bf16_t* __restrict__ xrcat, const int* __restrict__ off,
    const int* __restrict__ elist, const float* __restrict__ invDR,
    const float* __restrict__ bias, float* __restrict__ xf,
    bf16_t* __restrict__ xb) {
  int n = blockIdx.x, tid = threadIdx.x;
  float acc[3] = {0.f, 0.f, 0.f};
  int beg = off[n], end = off[n + 1];
  for (int j = beg; j < end; ++j) {
    int s = elist[j];
    int src = s >> 2, t = s & 3;
    float w = invDR[n * kR + t];
    const bf16_t* row = xrcat + (size_t)src * kNCAT + t * kD;
#pragma unroll
    for (int q = 0; q < 3; ++q) acc[q] += (float)row[tid + q * 256] * w;
  }
  const bf16_t* root = xrcat + (size_t)n * kNCAT + kR * kD;
#pragma unroll
  for (int q = 0; q < 3; ++q) {
    int d = tid + q * 256;
    float v = fmaxf(acc[q] + (float)root[d] + bias[d], 0.f);
    if (xf) xf[(size_t)n * kD + d] = v;
    if (xb) xb[(size_t)n * kD + d] = (bf16_t)v;
  }
}

// mean/max pool over 32 words; build ev (320,1536) f32 and cat320 (320,2304) bf16
__global__ void k_pool(const float* __restrict__ x2, const float* __restrict__ tok,
                       const int* __restrict__ csi, float* __restrict__ ev,
                       bf16_t* __restrict__ cat) {
  int s = blockIdx.x, tid = threadIdx.x;
  int b = s / 5;
  size_t cls = (size_t)csi[b] * 256 * kD;  // token_feats[seq, 0, :]
  for (int d = tid; d < kD; d += 256) {
    float mx = -1e30f, sum = 0.f;
    for (int w = 0; w < kWORDS; ++w) {
      float v = x2[((size_t)s * kWORDS + w) * kD + d];
      sum += v;
      mx = fmaxf(mx, v);
    }
    float mean = sum * (1.f / 32.f);
    ev[(size_t)s * 1536 + d] = mean;
    ev[(size_t)s * 1536 + kD + d] = mx;
    cat[(size_t)s * kDCAT + d] = (bf16_t)tok[cls + d];
    cat[(size_t)s * kDCAT + kD + d] = (bf16_t)mean;
    cat[(size_t)s * kDCAT + 2 * kD + d] = (bf16_t)mx;
  }
}

// p[s] = h[s,:] . att_w1 ; also attention output = sigmoid(p)
__global__ void k_p(const float* __restrict__ h, const float* __restrict__ w1,
                    float* __restrict__ p, float* __restrict__ out) {
  __shared__ float sh[256];
  int s = blockIdx.x, tid = threadIdx.x;
  float a = 0.f;
  for (int d = tid; d < kD; d += 256) a += h[(size_t)s * kD + d] * w1[d];
  sh[tid] = a;
  __syncthreads();
  for (int st = 128; st > 0; st >>= 1) {
    if (tid < st) sh[tid] += sh[tid + st];
    __syncthreads();
  }
  if (tid == 0) {
    p[s] = sh[0];
    out[kB * 3 + s] = 1.f / (1.f + expf(-sh[0]));  // attention[320]
  }
}

__global__ __launch_bounds__(256) void k_head(
    const float* __restrict__ ev, const float* __restrict__ ccls,
    const float* __restrict__ l1w, const float* __restrict__ l1b,
    const float* __restrict__ l2w, const float* __restrict__ l2b,
    const float* __restrict__ p, float* __restrict__ out) {
  __shared__ float rep[kDCAT];
  __shared__ float h1[1024];
  __shared__ float aw[5];
  __shared__ float red[256];
  __shared__ float osum[3];
  int b = blockIdx.x, tid = threadIdx.x;
  if (tid == 0) {
    float pv[5], pm = -1e30f;
    for (int e = 0; e < 5; ++e) { pv[e] = p[b * 5 + e]; pm = fmaxf(pm, pv[e]); }
    float s = 0.f;
    for (int e = 0; e < 5; ++e) { pv[e] = expf(pv[e] - pm); s += pv[e]; }
    for (int e = 0; e < 5; ++e) aw[e] = pv[e] / s;
  }
  __syncthreads();
  for (int d = tid; d < 1536; d += 256) {
    float g = 0.f;
    for (int e = 0; e < 5; ++e) g += aw[e] * ev[(size_t)(b * 5 + e) * 1536 + d];
    rep[d] = g;
  }
  for (int d = tid; d < kD; d += 256) rep[1536 + d] = ccls[(size_t)b * kD + d];
  __syncthreads();
  // lin1: each thread computes 4 outputs, coalesced column reads
  float a0 = l1b[tid], a1 = l1b[tid + 256], a2 = l1b[tid + 512], a3 = l1b[tid + 768];
  for (int d = 0; d < kDCAT; ++d) {
    float rv = rep[d];
    const float* wr = l1w + (size_t)d * 1024;
    a0 += rv * wr[tid];
    a1 += rv * wr[tid + 256];
    a2 += rv * wr[tid + 512];
    a3 += rv * wr[tid + 768];
  }
  h1[tid] = fmaxf(a0, 0.f);
  h1[tid + 256] = fmaxf(a1, 0.f);
  h1[tid + 512] = fmaxf(a2, 0.f);
  h1[tid + 768] = fmaxf(a3, 0.f);
  __syncthreads();
  float accv[3] = {0.f, 0.f, 0.f};
  for (int j = tid; j < 1024; j += 256) {
    float v = h1[j];
    accv[0] += v * l2w[j * 3 + 0];
    accv[1] += v * l2w[j * 3 + 1];
    accv[2] += v * l2w[j * 3 + 2];
  }
  for (int c = 0; c < 3; ++c) {
    red[tid] = accv[c];
    __syncthreads();
    for (int st = 128; st > 0; st >>= 1) {
      if (tid < st) red[tid] += red[tid + st];
      __syncthreads();
    }
    if (tid == 0) osum[c] = red[0] + l2b[c];
    __syncthreads();
  }
  if (tid == 0) {
    float m = fmaxf(osum[0], fmaxf(osum[1], osum[2]));
    float l = logf(expf(osum[0] - m) + expf(osum[1] - m) + expf(osum[2] - m));
    for (int c = 0; c < 3; ++c) out[b * 3 + c] = osum[c] - m - l;
  }
}

// ---------------------------------------------------------------------------
extern "C" void kernel_launch(void* const* d_in, const int* in_sizes, int n_in,
                              void* d_out, int out_size, void* d_ws, size_t ws_size,
                              hipStream_t stream) {
  (void)in_sizes; (void)n_in; (void)out_size; (void)ws_size;
  const float* tok    = (const float*)d_in[0];
  const float* ccls   = (const float*)d_in[1];
  const float* Wrel1  = (const float*)d_in[2];
  const float* Wroot1 = (const float*)d_in[3];
  const float* b1     = (const float*)d_in[4];
  const float* Wrel2  = (const float*)d_in[5];
  const float* Wroot2 = (const float*)d_in[6];
  const float* b2     = (const float*)d_in[7];
  const float* attw0  = (const float*)d_in[8];
  const float* attw1  = (const float*)d_in[9];
  const float* l1w    = (const float*)d_in[10];
  const float* l1b    = (const float*)d_in[11];
  const float* l2w    = (const float*)d_in[12];
  const float* l2b    = (const float*)d_in[13];
  const int*   wti    = (const int*)d_in[14];
  const int*   ei     = (const int*)d_in[15];
  const int*   et     = (const int*)d_in[16];
  const int*   csi    = (const int*)d_in[17];
  float* out = (float*)d_out;

  // workspace carve-out (256B aligned)
  char* w = (char*)d_ws;
  size_t o = 0;
  auto alloc = [&](size_t bytes) {
    void* p = w + o;
    o = (o + bytes + 255) & ~(size_t)255;
    return p;
  };
  bf16_t* xb      = (bf16_t*)alloc((size_t)kN * kD * 2);
  bf16_t* wcat1   = (bf16_t*)alloc((size_t)kD * kNCAT * 2);
  bf16_t* wcat2   = (bf16_t*)alloc((size_t)kD * kNCAT * 2);
  bf16_t* w0b     = (bf16_t*)alloc((size_t)kDCAT * kD * 2);
  bf16_t* xrcat   = (bf16_t*)alloc((size_t)kN * kNCAT * 2);
  float*  x2      = (float*)alloc((size_t)kN * kD * 4);
  int*    deg     = (int*)alloc((size_t)kN * 4);
  int*    off     = (int*)alloc((size_t)(kN + 1) * 4);
  int*    cursor  = (int*)alloc((size_t)kN * 4);
  int*    cntDR   = (int*)alloc((size_t)kN * kR * 4);
  float*  invDR   = (float*)alloc((size_t)kN * kR * 4);
  int*    elist   = (int*)alloc((size_t)kE * 4);
  float*  ev      = (float*)alloc((size_t)kSEQ * 1536 * 4);
  bf16_t* cat320b = (bf16_t*)alloc((size_t)kSEQ * kDCAT * 2);
  float*  hbuf    = (float*)alloc((size_t)kSEQ * kD * 4);
  float*  pvec    = (float*)alloc((size_t)kSEQ * 4);

  hipMemsetAsync(deg, 0, (size_t)kN * 4, stream);
  hipMemsetAsync(cntDR, 0, (size_t)kN * kR * 4, stream);

  // feature gather + weight conversion
  k_gather_mean<<<kN, 256, 0, stream>>>(tok, wti, xb);
  {
    int total = kD * kNCAT;
    k_pack_wcat<<<(total + 255) / 256, 256, 0, stream>>>(Wrel1, Wroot1, wcat1, total);
    k_pack_wcat<<<(total + 255) / 256, 256, 0, stream>>>(Wrel2, Wroot2, wcat2, total);
    int t2 = kDCAT * kD;
    k_f32_to_bf16<<<(t2 + 255) / 256, 256, 0, stream>>>(attw0, w0b, t2);
  }
  // CSR build
  k_count<<<(kE + 255) / 256, 256, 0, stream>>>(ei, et, deg, cntDR);
  k_scan<<<1, 1024, 0, stream>>>(deg, off, cursor);
  k_fill<<<(kE + 255) / 256, 256, 0, stream>>>(ei, et, cursor, elist);
  k_inv<<<(kN * kR + 255) / 256, 256, 0, stream>>>(cntDR, invDR);

  dim3 gBig(kN / 128, kNCAT / 128);
  // layer 1
  k_gemm_bf16<<<gBig, 256, 0, stream>>>(xb, wcat1, nullptr, xrcat, kN, kNCAT, kD, 0);
  k_aggregate<<<kN, 256, 0, stream>>>(xrcat, off, elist, invDR, b1, nullptr, xb);
  // layer 2
  k_gemm_bf16<<<gBig, 256, 0, stream>>>(xb, wcat2, nullptr, xrcat, kN, kNCAT, kD, 0);
  k_aggregate<<<kN, 256, 0, stream>>>(xrcat, off, elist, invDR, b2, x2, nullptr);
  // pooling + attention
  k_pool<<<kSEQ, 256, 0, stream>>>(x2, tok, csi, ev, cat320b);
  dim3 gAtt((kSEQ + 127) / 128, kD / 128);
  k_gemm_bf16<<<gAtt, 256, 0, stream>>>(cat320b, w0b, hbuf, nullptr, kSEQ, kD, kDCAT, 1);
  k_p<<<kSEQ, 256, 0, stream>>>(hbuf, attw1, pvec, out);
  // head
  k_head<<<kB, 256, 0, stream>>>(ev, ccls, l1w, l1b, l2w, l2b, pvec, out);
}
